// Encoder_66468913873581
// MI455X (gfx1250) — matrix operand order repaired and checked by reading, not compiled, not run on hardware
//
#include <hip/hip_runtime.h>
#include <hip/hip_bf16.h>
#include <math.h>

// ---------------------------------------------------------------------------
// Types for CDNA5 WMMA (wave32): D(16x16 f32) = A(16x32 bf16) x B(32x16 bf16) + C
// ---------------------------------------------------------------------------
typedef __bf16 bf16_t;
typedef __attribute__((ext_vector_type(16))) __bf16 v16bf;
typedef __attribute__((ext_vector_type(8)))  __bf16 v8bf;
typedef __attribute__((ext_vector_type(8)))  float  v8f;

static constexpr int DMODEL = 1024;
static constexpr int SEQ    = 2048;
static constexpr int NBATCH = 4;
static constexpr int BS     = NBATCH * SEQ;      // 8192 total rows
static constexpr float SM_SCALE = 0.03125f;      // 1/sqrt(1024)

__device__ __forceinline__ v8f wmma_bf16(v16bf a, v16bf b, v8f c) {
  // (neg_a, A, neg_b, B, c_mod, C, reuse_a, reuse_b)
  return __builtin_amdgcn_wmma_f32_16x16x32_bf16(false, a, false, b, (short)0, c,
                                                 false, false);
}

// A fragment of a row-major matrix A[.][lda] : 16x32 tile at (m0, k0).
// ISA layout (16-bit A 16x32): lane l holds M = m0+(l&15);
// element e maps to K = k0 + 16*(e>>3) + 8*(l>>4) + (e&7)  -> two contiguous
// 8-element (16B) runs per lane.
__device__ __forceinline__ v16bf load_a(const bf16_t* A, int lda, int m0, int k0,
                                        int lane) {
  const int h = lane >> 4;
  const bf16_t* row = A + (size_t)(m0 + (lane & 15)) * lda + k0 + 8 * h;
  v8bf lo = *(const v8bf*)row;
  v8bf hi = *(const v8bf*)(row + 16);
  v16bf r;
#pragma unroll
  for (int i = 0; i < 8; ++i) { r[i] = lo[i]; r[i + 8] = hi[i]; }
  return r;
}

// B fragment from K-contiguous storage Bt[n][k] (ld = ldb): 32x16 (KxN) tile
// at (k0, n0). ISA layout: lane l holds N = n0+(l&15); element e -> K =
// k0 + 16*(e>>3) + 8*(l>>4) + (e&7)  -> two contiguous 8-element (16B) runs per lane.
__device__ __forceinline__ v16bf load_b(const bf16_t* Bt, int ldb, int n0, int k0,
                                        int lane) {
  const bf16_t* p = Bt + (size_t)(n0 + (lane & 15)) * ldb + k0 + 8 * (lane >> 4);
  v8bf lo = *(const v8bf*)p;
  v8bf hi = *(const v8bf*)(p + 16);
  v16bf r;
#pragma unroll
  for (int i = 0; i < 8; ++i) { r[i] = lo[i]; r[i + 8] = hi[i]; }
  return r;
}

// ---------------------------------------------------------------------------
// CDNA5 async copy: global -> LDS (16 bytes), tracked by ASYNCcnt.
// LDS byte address in the vdst VGPR; vaddr = 64-bit global address (GV mode).
// ---------------------------------------------------------------------------
__device__ __forceinline__ void async_cp16(unsigned int lds_byte_addr,
                                           const void* gaddr) {
  asm volatile("global_load_async_to_lds_b128 %0, %1, off"
               :: "v"(lds_byte_addr), "v"((unsigned long long)gaddr)
               : "memory");
}
__device__ __forceinline__ void wait_async3() {
  asm volatile("s_wait_asynccnt 0x3" ::: "memory");
}
__device__ __forceinline__ void wait_async0() {
  asm volatile("s_wait_asynccnt 0x0" ::: "memory");
}

// ---------------------------------------------------------------------------
// Kernel 0a: x (f32) -> xb (bf16), 4 elements/thread
// ---------------------------------------------------------------------------
__global__ void k_cvt_x(const float* __restrict__ x, bf16_t* __restrict__ xb) {
  int i = (blockIdx.x * 256 + threadIdx.x) * 4;
  float4 v = *(const float4*)(x + i);
  bf16_t* p = xb + i;
  p[0] = (__bf16)v.x; p[1] = (__bf16)v.y; p[2] = (__bf16)v.z; p[3] = (__bf16)v.w;
}

// ---------------------------------------------------------------------------
// Kernel 0b: W[k][n] (f32, 1024x1024) -> Wt[n][k] (bf16)  (LDS tiled transpose)
// ---------------------------------------------------------------------------
__global__ void k_wt(const float* __restrict__ W, bf16_t* __restrict__ Wt) {
  __shared__ float t[32][33];
  const int bn = blockIdx.x * 32;   // n block
  const int bk = blockIdx.y * 32;   // k block
  const int tx = threadIdx.x, ty = threadIdx.y;   // (32, 8)
#pragma unroll
  for (int i = 0; i < 32; i += 8)
    t[ty + i][tx] = W[(size_t)(bk + ty + i) * DMODEL + bn + tx];
  __syncthreads();
#pragma unroll
  for (int i = 0; i < 32; i += 8)
    Wt[(size_t)(bn + ty + i) * DMODEL + bk + tx] = (__bf16)t[tx][ty + i];
}

// ---------------------------------------------------------------------------
// Kernel 1: QKV projections with async-LDS double buffering.
// C[8192 x 1024] = xb @ W for z in {Q,K,V}.
// Workgroup: 8 waves as 4(M)x2(N) -> tile 128x64; wave tile 32x32 (4 WMMA acc).
// Per 32-deep K chunk: A tile 128x32 (padded rows of 40 elems), B tile 64x32
// (padded rows of 40), filled with global_load_async_to_lds_b128 and consumed
// from LDS. Q,K stored row-major bf16; V stored transposed Vt[b][dv][s].
// ---------------------------------------------------------------------------
static constexpr int LDA_T   = 40;                  // padded row (elements)
static constexpr int A_ELEMS = 128 * LDA_T;         // 5120
static constexpr int B_ELEMS = 64 * LDA_T;          // 2560
static constexpr int A0_OFF  = 0;
static constexpr int A1_OFF  = A_ELEMS;             // 5120
static constexpr int B0_OFF  = 2 * A_ELEMS;         // 10240
static constexpr int B1_OFF  = 2 * A_ELEMS + B_ELEMS;

__global__ __launch_bounds__(256) void
k_qkv(const bf16_t* __restrict__ xb,
      const bf16_t* __restrict__ Wqt, const bf16_t* __restrict__ Wkt,
      const bf16_t* __restrict__ Wvt,
      bf16_t* __restrict__ Qm, bf16_t* __restrict__ Km, bf16_t* __restrict__ Vt) {
  __shared__ bf16_t smem[2 * A_ELEMS + 2 * B_ELEMS];

  const int tid = threadIdx.x;
  const int lane = tid & 31, wv = tid >> 5, half = lane >> 4;
  const int wm = wv & 3, wn = wv >> 2;
  const int M0 = blockIdx.x * 128;
  const int N0 = blockIdx.y * 64;
  const int mw_l = wm * 32, nw_l = wn * 32;   // wave origin inside LDS tiles
  const int z = blockIdx.z;
  const bf16_t* Wt = (z == 0) ? Wqt : (z == 1) ? Wkt : Wvt;

  const unsigned int smem_base = (unsigned int)(uintptr_t)(&smem[0]);

  // per-thread staging assignments (3 async b128 issues per chunk)
  const int arow0 = tid >> 2,          asub0 = tid & 3;           // A chunk 0
  const int arow1 = (tid + 256) >> 2,  asub1 = tid & 3;           // A chunk 1
  const int brow  = tid >> 2,          bsub  = tid & 3;           // B chunk

  auto stage = [&](int k0, int buf) {
    const unsigned int aBase = smem_base + (buf ? A1_OFF : A0_OFF) * 2;
    const unsigned int bBase = smem_base + (buf ? B1_OFF : B0_OFF) * 2;
    async_cp16(aBase + (unsigned)(arow0 * LDA_T + asub0 * 8) * 2,
               xb + (size_t)(M0 + arow0) * DMODEL + k0 + asub0 * 8);
    async_cp16(aBase + (unsigned)(arow1 * LDA_T + asub1 * 8) * 2,
               xb + (size_t)(M0 + arow1) * DMODEL + k0 + asub1 * 8);
    async_cp16(bBase + (unsigned)(brow * LDA_T + bsub * 8) * 2,
               Wt + (size_t)(N0 + brow) * DMODEL + k0 + bsub * 8);
  };

  v8f zero = {};
  v8f acc[2][2];
#pragma unroll
  for (int i = 0; i < 2; ++i)
#pragma unroll
    for (int j = 0; j < 2; ++j) acc[i][j] = zero;

  const int NIT = DMODEL / 32;   // 32 chunks
  stage(0, 0);
  for (int it = 0; it < NIT; ++it) {
    const int cur = it & 1;
    if (it + 1 < NIT) {
      stage((it + 1) * 32, (it + 1) & 1);
      wait_async3();            // the 3 newest (next chunk) may stay in flight
    } else {
      wait_async0();
    }
    __syncthreads();            // chunk `it` visible to the whole workgroup

    const bf16_t* At = smem + (cur ? A1_OFF : A0_OFF);
    const bf16_t* Bt = smem + (cur ? B1_OFF : B0_OFF);
    v16bf a0 = load_a(At, LDA_T, mw_l,      0, lane);
    v16bf a1 = load_a(At, LDA_T, mw_l + 16, 0, lane);
    v16bf b0 = load_b(Bt, LDA_T, nw_l,      0, lane);
    v16bf b1 = load_b(Bt, LDA_T, nw_l + 16, 0, lane);
    acc[0][0] = wmma_bf16(a0, b0, acc[0][0]);
    acc[0][1] = wmma_bf16(a0, b1, acc[0][1]);
    acc[1][0] = wmma_bf16(a1, b0, acc[1][0]);
    acc[1][1] = wmma_bf16(a1, b1, acc[1][1]);

    __syncthreads();            // everyone done reading buf `cur`
  }

  const int mw = M0 + mw_l;
  const int nw = N0 + nw_l;
  if (z < 2) {
    bf16_t* Out = z ? Km : Qm;
#pragma unroll
    for (int ms = 0; ms < 2; ++ms)
#pragma unroll
      for (int ns = 0; ns < 2; ++ns) {
        const int n = nw + ns * 16 + (lane & 15);
#pragma unroll
        for (int r = 0; r < 8; ++r) {
          const int m = mw + ms * 16 + 8 * half + r;
          Out[(size_t)m * DMODEL + n] = (__bf16)acc[ms][ns][r];
        }
      }
  } else {
    // V transposed store: Vt[b][n][s]; D-layout makes 8 rows contiguous.
#pragma unroll
    for (int ms = 0; ms < 2; ++ms)
#pragma unroll
      for (int ns = 0; ns < 2; ++ns) {
        const int n = nw + ns * 16 + (lane & 15);
        const int mbase = mw + ms * 16 + 8 * half;      // multiple of 8
        const int b = mbase >> 11, s = mbase & (SEQ - 1);
        v8bf pk;
#pragma unroll
        for (int r = 0; r < 8; ++r) pk[r] = (__bf16)acc[ms][ns][r];
        *(v8bf*)(Vt + ((size_t)b * DMODEL + n) * SEQ + s) = pk;
      }
  }
}

// ---------------------------------------------------------------------------
// Kernel 2: flash attention.  Br=32 query rows / block, Bc=64 keys / step.
// 8 waves: scores -> one 16x16 subtile per wave (2x4); O accum: each wave owns
// a 128-wide Dv slice = 2x8 WMMA tiles (128 VGPRs).
// ---------------------------------------------------------------------------
__global__ __launch_bounds__(256) void
k_attn(const bf16_t* __restrict__ Qm, const bf16_t* __restrict__ Km,
       const bf16_t* __restrict__ Vt, bf16_t* __restrict__ ctx) {
  __shared__ float  Sld[32][68];
  __shared__ bf16_t Pld[32][80];
  __shared__ float  mrun[32], lrun[32], cfac[32];

  const int tid = threadIdx.x, lane = tid & 31, wv = tid >> 5;
  const int half = lane >> 4, l15 = lane & 15;
  const int b = blockIdx.y, q0 = blockIdx.x * 32;
  const bf16_t* Qb = Qm + (size_t)b * SEQ * DMODEL;
  const bf16_t* Kb = Km + (size_t)b * SEQ * DMODEL;
  const bf16_t* Vb = Vt + (size_t)b * DMODEL * SEQ;

  if (tid < 32) { mrun[tid] = -INFINITY; lrun[tid] = 0.f; }

  v8f zero = {};
  v8f o[2][8];
#pragma unroll
  for (int i = 0; i < 2; ++i)
#pragma unroll
    for (int j = 0; j < 8; ++j) o[i][j] = zero;

  const int msub = wv & 1, nsub = wv >> 1;    // score subtile of this wave
  __syncthreads();

  for (int kb = 0; kb < SEQ; kb += 64) {
    // prefetch next key tile rows + next V K-slice (global_prefetch_b8)
    if (kb + 64 < SEQ) {
      __builtin_prefetch(Kb + (size_t)(kb + 64 + (tid >> 3)) * DMODEL, 0, 1);
      __builtin_prefetch(Vb + (size_t)(wv * 128 + (lane >> 2)) * SEQ + kb + 64, 0, 1);
    }

    // ---- scores subtile: S = Q Kt  (K row-major == K^T's B operand) ----
    v8f s = zero;
    for (int k0 = 0; k0 < DMODEL; k0 += 32) {
      v16bf a  = load_a(Qb, DMODEL, q0 + msub * 16, k0, lane);
      v16bf bb = load_b(Kb, DMODEL, kb + nsub * 16, k0, lane);
      s = wmma_bf16(a, bb, s);
    }
#pragma unroll
    for (int r = 0; r < 8; ++r)
      Sld[msub * 16 + 8 * half + r][nsub * 16 + l15] = s[r];
    __syncthreads();

    // ---- online softmax: 8 threads per row, 8 columns each ----
    {
      const int row = tid >> 3, c8 = tid & 7;
      float v[8]; float mx = -INFINITY;
#pragma unroll
      for (int i = 0; i < 8; ++i) {
        v[i] = Sld[row][c8 * 8 + i] * SM_SCALE;
        mx = fmaxf(mx, v[i]);
      }
      mx = fmaxf(mx, __shfl_xor(mx, 1, 32));
      mx = fmaxf(mx, __shfl_xor(mx, 2, 32));
      mx = fmaxf(mx, __shfl_xor(mx, 4, 32));
      const float mold = mrun[row];
      const float mnew = fmaxf(mold, mx);
      const float c    = __expf(mold - mnew);
      float ls = 0.f; v8bf pv;
#pragma unroll
      for (int i = 0; i < 8; ++i) {
        float p = __expf(v[i] - mnew);
        ls += p; pv[i] = (__bf16)p;
      }
      ls += __shfl_xor(ls, 1, 32);
      ls += __shfl_xor(ls, 2, 32);
      ls += __shfl_xor(ls, 4, 32);
      *(v8bf*)&Pld[row][c8 * 8] = pv;
      if (c8 == 0) { lrun[row] = lrun[row] * c + ls; mrun[row] = mnew; cfac[row] = c; }
    }
    __syncthreads();

    // ---- rescale O accumulators by per-row correction ----
#pragma unroll
    for (int ms = 0; ms < 2; ++ms) {
      float cv[8];
#pragma unroll
      for (int r = 0; r < 8; ++r) cv[r] = cfac[ms * 16 + 8 * half + r];
#pragma unroll
      for (int ns = 0; ns < 8; ++ns)
#pragma unroll
        for (int r = 0; r < 8; ++r) o[ms][ns][r] *= cv[r];
    }

    // ---- O += P @ V  (P from LDS as A; Vt K-contiguous as B) ----
    for (int kc = 0; kc < 64; kc += 32) {
      v16bf a0 = load_a(&Pld[0][0], 80, 0,  kc, lane);
      v16bf a1 = load_a(&Pld[0][0], 80, 16, kc, lane);
#pragma unroll
      for (int ns = 0; ns < 8; ++ns) {
        v16bf bb = load_b(Vb, SEQ, wv * 128 + ns * 16, kb + kc, lane);
        o[0][ns] = wmma_bf16(a0, bb, o[0][ns]);
        o[1][ns] = wmma_bf16(a1, bb, o[1][ns]);
      }
    }
    __syncthreads();
  }

  if (tid < 32) cfac[tid] = 1.0f / lrun[tid];
  __syncthreads();

#pragma unroll
  for (int ms = 0; ms < 2; ++ms) {
    float cv[8];
#pragma unroll
    for (int r = 0; r < 8; ++r) cv[r] = cfac[ms * 16 + 8 * half + r];
#pragma unroll
    for (int ns = 0; ns < 8; ++ns) {
      const int n = wv * 128 + ns * 16 + l15;
#pragma unroll
      for (int r = 0; r < 8; ++r) {
        const int m = q0 + ms * 16 + 8 * half + r;
        ctx[((size_t)b * SEQ + m) * DMODEL + n] = (__bf16)(o[ms][ns][r] * cv[r]);
      }
    }
  }
}

// ---------------------------------------------------------------------------
// Kernel 3: hidden = LayerNorm(ctx @ Wo).  Block = 32 rows x 1024 cols,
// 8 waves each own a 128-col slice (2x8 WMMA acc). Row stats via shuffle +
// ds_add_f32 atomics.
// ---------------------------------------------------------------------------
__global__ __launch_bounds__(256) void
k_oproj(const bf16_t* __restrict__ ctx, const bf16_t* __restrict__ Wot,
        const float* __restrict__ gamma, const float* __restrict__ beta,
        float* __restrict__ out) {
  __shared__ float rsum[32], rsq[32], mean_s[32], rstd_s[32];
  const int tid = threadIdx.x, lane = tid & 31, wv = tid >> 5;
  const int half = lane >> 4, l15 = lane & 15;
  const int m0 = blockIdx.x * 32;

  if (tid < 32) { rsum[tid] = 0.f; rsq[tid] = 0.f; }

  v8f zero = {};
  v8f o[2][8];
#pragma unroll
  for (int i = 0; i < 2; ++i)
#pragma unroll
    for (int j = 0; j < 8; ++j) o[i][j] = zero;
  __syncthreads();

  for (int k0 = 0; k0 < DMODEL; k0 += 32) {
    v16bf a0 = load_a(ctx, DMODEL, m0,      k0, lane);
    v16bf a1 = load_a(ctx, DMODEL, m0 + 16, k0, lane);
#pragma unroll
    for (int ns = 0; ns < 8; ++ns) {
      v16bf bb = load_b(Wot, DMODEL, wv * 128 + ns * 16, k0, lane);
      o[0][ns] = wmma_bf16(a0, bb, o[0][ns]);
      o[1][ns] = wmma_bf16(a1, bb, o[1][ns]);
    }
  }

  // row sums / sumsq across this wave's 128 columns, then LDS atomics
#pragma unroll
  for (int ms = 0; ms < 2; ++ms)
#pragma unroll
    for (int r = 0; r < 8; ++r) {
      float s = 0.f, q = 0.f;
#pragma unroll
      for (int ns = 0; ns < 8; ++ns) {
        float v = o[ms][ns][r];
        s += v; q += v * v;
      }
      s += __shfl_xor(s, 1, 32);  q += __shfl_xor(q, 1, 32);
      s += __shfl_xor(s, 2, 32);  q += __shfl_xor(q, 2, 32);
      s += __shfl_xor(s, 4, 32);  q += __shfl_xor(q, 4, 32);
      s += __shfl_xor(s, 8, 32);  q += __shfl_xor(q, 8, 32);
      if (l15 == 0) {
        atomicAdd(&rsum[ms * 16 + 8 * half + r], s);
        atomicAdd(&rsq [ms * 16 + 8 * half + r], q);
      }
    }
  __syncthreads();

  if (tid < 32) {
    const float mu  = rsum[tid] * (1.0f / DMODEL);
    const float var = rsq[tid] * (1.0f / DMODEL) - mu * mu;
    mean_s[tid] = mu;
    rstd_s[tid] = rsqrtf(var + 1e-5f);
  }
  __syncthreads();

#pragma unroll
  for (int ms = 0; ms < 2; ++ms)
#pragma unroll
    for (int ns = 0; ns < 8; ++ns) {
      const int n = wv * 128 + ns * 16 + l15;
      const float g = gamma[n], be = beta[n];
#pragma unroll
      for (int r = 0; r < 8; ++r) {
        const int row = ms * 16 + 8 * half + r;
        const int m = m0 + row;
        out[(size_t)m * DMODEL + n] =
            (o[ms][ns][r] - mean_s[row]) * rstd_s[row] * g + be;
      }
    }
}

// ---------------------------------------------------------------------------
// Host launcher.  Workspace layout (bytes):
//   xb   @ 0       16 MB  (8192x1024 bf16)
//   Wqt  @ 16 MB    2 MB
//   Wkt  @ 18 MB    2 MB
//   Wvt  @ 20 MB    2 MB
//   Wot  @ 22 MB    2 MB
//   Q    @ 24 MB   16 MB
//   K    @ 40 MB   16 MB
//   Vt   @ 56 MB   16 MB
//   ctx  @ 72 MB   16 MB   => 88 MB total
// ---------------------------------------------------------------------------
extern "C" void kernel_launch(void* const* d_in, const int* in_sizes, int n_in,
                              void* d_out, int out_size, void* d_ws, size_t ws_size,
                              hipStream_t stream) {
  const float* x     = (const float*)d_in[0];
  const float* Wq    = (const float*)d_in[1];
  const float* Wk    = (const float*)d_in[2];
  const float* Wv    = (const float*)d_in[3];
  const float* Wo    = (const float*)d_in[4];
  const float* gamma = (const float*)d_in[5];
  const float* beta  = (const float*)d_in[6];
  float* out = (float*)d_out;

  char* ws = (char*)d_ws;
  const size_t MB = 1024ull * 1024ull;
  bf16_t* xb  = (bf16_t*)(ws + 0);
  bf16_t* Wqt = (bf16_t*)(ws + 16 * MB);
  bf16_t* Wkt = (bf16_t*)(ws + 18 * MB);
  bf16_t* Wvt = (bf16_t*)(ws + 20 * MB);
  bf16_t* Wot = (bf16_t*)(ws + 22 * MB);
  bf16_t* Qm  = (bf16_t*)(ws + 24 * MB);
  bf16_t* Km  = (bf16_t*)(ws + 40 * MB);
  bf16_t* Vt  = (bf16_t*)(ws + 56 * MB);
  bf16_t* ctx = (bf16_t*)(ws + 72 * MB);

  // 0a: convert activations to bf16
  k_cvt_x<<<BS * DMODEL / (256 * 4), 256, 0, stream>>>(x, xb);
  // 0b: transpose+convert the four weight matrices to [out][in] bf16
  dim3 tb(32, 8), tg(32, 32);
  k_wt<<<tg, tb, 0, stream>>>(Wq, Wqt);
  k_wt<<<tg, tb, 0, stream>>>(Wk, Wkt);
  k_wt<<<tg, tb, 0, stream>>>(Wv, Wvt);
  k_wt<<<tg, tb, 0, stream>>>(Wo, Wot);
  // 1: QKV projections (z = 0/1/2), async-LDS double buffered
  k_qkv<<<dim3(BS / 128, DMODEL / 64, 3), 256, 0, stream>>>(xb, Wqt, Wkt, Wvt,
                                                            Qm, Km, Vt);
  // 2: flash attention
  k_attn<<<dim3(SEQ / 32, NBATCH), 256, 0, stream>>>(Qm, Km, Vt, ctx);
  // 3: output projection + layernorm
  k_oproj<<<BS / 32, 256, 0, stream>>>(ctx, Wot, gamma, beta, out);
}
